// Attention_19301583028761
// MI455X (gfx1250) — compile-verified, matrix operand order checked by reading, not run on hardware
//
#include <hip/hip_runtime.h>
#include <hip/hip_bf16.h>
#include <math.h>

// ---------------- problem constants ----------------
#define BB    8
#define CC    192
#define HH    128
#define WW    128
#define HWSZ  (HH * WW)          // 16384
#define NHD   6
#define HD    32
#define WS    8
#define NTOK  64                 // WS*WS
#define NWIN  2048               // B * (H/8) * (W/8)
#define SCALE_C 0.17677669529663688f   // 32^-0.5

typedef __attribute__((ext_vector_type(16))) __bf16 v16bf;
typedef __attribute__((ext_vector_type(8)))  __bf16 v8bf;
typedef __attribute__((ext_vector_type(8)))  float  v8f;
typedef __attribute__((ext_vector_type(2)))  float  v2f;

// 16-bit WMMA fragment (16x32): lane's 16 values are two contiguous 8-elem runs.
// elements 0..7  -> k = kk + 8*half + 0..7
// elements 8..15 -> k = kk + 16 + 8*half + 0..7
__device__ __forceinline__ v16bf load_frag_bf16(const __bf16* __restrict__ row,
                                                int kk, int half) {
    v8bf lo = *reinterpret_cast<const v8bf*>(row + kk + 8 * half);
    v8bf hi = *reinterpret_cast<const v8bf*>(row + kk + 16 + 8 * half);
    return __builtin_shufflevector(lo, hi, 0, 1, 2, 3, 4, 5, 6, 7,
                                   8, 9, 10, 11, 12, 13, 14, 15);
}

// ---------------- kernel 0: meta-MLP relative position bias ----------------
__global__ __launch_bounds__(256)
void meta_bias_kernel(const float* __restrict__ w1, const float* __restrict__ b1,
                      const float* __restrict__ w2, const float* __restrict__ b2,
                      float* __restrict__ bias) {
    int idx = blockIdx.x * blockDim.x + threadIdx.x;
    if (idx >= NTOK * NTOK) return;
    int p = idx >> 6, q = idx & 63;
    int i1 = p >> 3, j1 = p & 7, i2 = q >> 3, j2 = q & 7;
    float r0 = (float)(i1 - i2), r1 = (float)(j1 - j2);
    float s0 = (r0 > 0.f) ? 1.f : ((r0 < 0.f) ? -1.f : 0.f);
    float s1 = (r1 > 0.f) ? 1.f : ((r1 < 0.f) ? -1.f : 0.f);
    r0 = s0 * log1pf(fabsf(r0));
    r1 = s1 * log1pf(fabsf(r1));
    float out[NHD];
    #pragma unroll
    for (int o = 0; o < NHD; ++o) out[o] = b2[o];
    for (int h = 0; h < 256; ++h) {
        float a = fmaf(r0, w1[h], fmaf(r1, w1[256 + h], b1[h]));
        a = fmaxf(a, 0.f);
        #pragma unroll
        for (int o = 0; o < NHD; ++o) out[o] = fmaf(a, w2[h * NHD + o], out[o]);
    }
    #pragma unroll
    for (int o = 0; o < NHD; ++o) bias[o * 4096 + p * 64 + q] = out[o];
}

// ---------------- kernel 1: QKV projection GEMM (bf16 WMMA, f32 accum) ----------------
// 64(M) x 64(N) block tile; wave = 16x32 strip (A-fragment reuse).
__global__ __launch_bounds__(256)
void qkv_gemm_kernel(const float* __restrict__ x,
                     const float* __restrict__ qkw, const float* __restrict__ qkb,
                     const float* __restrict__ vw,  const float* __restrict__ vb,
                     float* __restrict__ Q, float* __restrict__ K, float* __restrict__ V) {
    __shared__ __bf16 As[64][CC + 8];   // [m][k]
    __shared__ __bf16 Bs[64][CC + 8];   // [n][k]  (row-major in K -> b128 fragment loads)

    const int m0  = blockIdx.x * 64;
    const int n0  = blockIdx.y * 64;    // 0..575, tiles never cross q/k/v segment bounds
    const int tid = threadIdx.x;
    const int b   = m0 >> 14;
    const int hw0 = m0 & (HWSZ - 1);

    // A tile: float4 along contiguous hw, transpose-store as bf16
    for (int i = tid; i < 64 * CC / 4; i += 256) {
        int c = i >> 4, m4 = (i & 15) * 4;
        float4 xv = *reinterpret_cast<const float4*>(
            &x[(size_t)(b * CC + c) * HWSZ + hw0 + m4]);
        As[m4 + 0][c] = (__bf16)xv.x;
        As[m4 + 1][c] = (__bf16)xv.y;
        As[m4 + 2][c] = (__bf16)xv.z;
        As[m4 + 3][c] = (__bf16)xv.w;
    }
    // B tile: 64 weight rows, contiguous in c
    for (int i = tid; i < 64 * CC / 4; i += 256) {
        int nl = i / 48, c4 = (i % 48) * 4;
        int oc = n0 + nl;
        const float* wr = (oc < 384) ? &qkw[oc * CC] : &vw[(oc - 384) * CC];
        float4 wv = *reinterpret_cast<const float4*>(wr + c4);
        Bs[nl][c4 + 0] = (__bf16)wv.x;
        Bs[nl][c4 + 1] = (__bf16)wv.y;
        Bs[nl][c4 + 2] = (__bf16)wv.z;
        Bs[nl][c4 + 3] = (__bf16)wv.w;
    }
    __syncthreads();

    const int wave = tid >> 5, lane = tid & 31;
    const int wm  = (wave & 3) * 16;
    const int wnb = (wave >> 2) * 32;
    const int half = lane >> 4, l15 = lane & 15;

    const __bf16* arow  = As[wm + l15];
    const __bf16* brow0 = Bs[wnb + l15];
    const __bf16* brow1 = Bs[wnb + 16 + l15];

    v8f acc0 = {}, acc1 = {};
    #pragma unroll
    for (int kk = 0; kk < CC; kk += 32) {
        v16bf af  = load_frag_bf16(arow,  kk, half);
        v16bf bf0 = load_frag_bf16(brow0, kk, half);
        v16bf bf1 = load_frag_bf16(brow1, kk, half);
        acc0 = __builtin_amdgcn_wmma_f32_16x16x32_bf16(false, af, false, bf0,
                                                       (short)0, acc0, false, false);
        acc1 = __builtin_amdgcn_wmma_f32_16x16x32_bf16(false, af, false, bf1,
                                                       (short)0, acc1, false, false);
    }

    // epilogue: bias, q-scale, scatter into window layout [wi][t][CC]
    #pragma unroll
    for (int s = 0; s < 2; ++s) {
        const v8f& acc = s ? acc1 : acc0;
        const int oc = n0 + wnb + 16 * s + l15;    // segment uniform per block
        #pragma unroll
        for (int i = 0; i < 8; ++i) {
            int m  = m0 + wm + i + 8 * half;
            int hw = m & (HWSZ - 1);
            int h = hw >> 7, w = hw & 127;
            int wi = (b * 16 + (h >> 3)) * 16 + (w >> 3);
            int t  = ((h & 7) << 3) | (w & 7);
            float val = acc[i];
            if (oc < 192) {
                val = (val + qkb[oc]) * SCALE_C;
                Q[(size_t)wi * NTOK * CC + t * CC + oc] = val;
            } else if (oc < 384) {
                val += qkb[oc];
                K[(size_t)wi * NTOK * CC + t * CC + (oc - 192)] = val;
            } else {
                val += vb[oc - 384];
                V[(size_t)wi * NTOK * CC + t * CC + (oc - 384)] = val;
            }
        }
    }
}

// ---------------- kernel 2: per-(window, head) attention ----------------
__global__ __launch_bounds__(128)
void attn_kernel(const float* __restrict__ Q, const float* __restrict__ K,
                 const float* __restrict__ V, const float* __restrict__ bias,
                 float* __restrict__ Y) {
    __shared__ float qs[NTOK][36], ks2[NTOK][36], vs[NTOK][36], os[NTOK][36];
    __shared__ float vsT[HD][66];       // transposed V for second matmul B-fragments
    __shared__ float at[NTOK][66];

    const int wi = blockIdx.x / NHD;
    const int hd = blockIdx.x % NHD;
    const size_t base = (size_t)wi * NTOK * CC + hd * HD;
    const int tid = threadIdx.x;

    for (int i = tid; i < NTOK * HD / 4; i += 128) {
        int t = i >> 3, c4 = (i & 7) * 4;
        float4 qv = *reinterpret_cast<const float4*>(&Q[base + t * CC + c4]);
        float4 kv = *reinterpret_cast<const float4*>(&K[base + t * CC + c4]);
        float4 vv = *reinterpret_cast<const float4*>(&V[base + t * CC + c4]);
        *reinterpret_cast<float4*>(&qs[t][c4])  = qv;
        *reinterpret_cast<float4*>(&ks2[t][c4]) = kv;
        *reinterpret_cast<float4*>(&vs[t][c4])  = vv;
        vsT[c4 + 0][t] = vv.x;
        vsT[c4 + 1][t] = vv.y;
        vsT[c4 + 2][t] = vv.z;
        vsT[c4 + 3][t] = vv.w;
    }
    __syncthreads();

    // 8x8 circular conv per channel: o[n,d] = sum q[(u,v),d] * k[((y-u)&7,(x-v)&7),d]
    for (int p = tid; p < NTOK * HD / 2; p += 128) {
        int n = p >> 4, d2 = (p & 15) * 2;
        int y = n >> 3, x = n & 7;
        v2f accv = {0.f, 0.f};
        #pragma unroll
        for (int u = 0; u < 8; ++u) {
            int ky = ((y - u) & 7) << 3;
            #pragma unroll
            for (int v = 0; v < 8; ++v) {
                v2f qv = *reinterpret_cast<const v2f*>(&qs[u * 8 + v][d2]);
                v2f kv = *reinterpret_cast<const v2f*>(&ks2[ky + ((x - v) & 7)][d2]);
                accv = qv * kv + accv;
            }
        }
        *reinterpret_cast<v2f*>(&os[n][d2]) = accv;
    }
    __syncthreads();

    const int wave = tid >> 5, lane = tid & 31;
    const int half = lane >> 4, col = lane & 15;
    const int tr = wave * 16;

    // attn = o . v^T  (f32 WMMA 16x16x4, K=HD)
    for (int tcb = 0; tcb < 4; ++tcb) {
        int tc = tcb * 16;
        v8f acc = {};
        #pragma unroll
        for (int kk = 0; kk < HD; kk += 4) {
            v2f af = *reinterpret_cast<const v2f*>(&os[tr + col][kk + 2 * half]);
            v2f bf = *reinterpret_cast<const v2f*>(&vs[tc + col][kk + 2 * half]);
            acc = __builtin_amdgcn_wmma_f32_16x16x4_f32(false, af, false, bf,
                                                        (short)0, acc, false, false);
        }
        #pragma unroll
        for (int i = 0; i < 8; ++i) {
            int n = tr + i + 8 * half, m = tc + col;
            at[n][m] = acc[i] + bias[hd * 4096 + n * 64 + m];
        }
    }
    __syncthreads();

    // row softmax
    if (tid < NTOK) {
        float mx = -1e30f;
        for (int m = 0; m < NTOK; ++m) mx = fmaxf(mx, at[tid][m]);
        float s = 0.f;
        for (int m = 0; m < NTOK; ++m) {
            float e = __expf(at[tid][m] - mx);
            at[tid][m] = e;
            s += e;
        }
        float inv = 1.f / s;
        for (int m = 0; m < NTOK; ++m) at[tid][m] *= inv;
    }
    __syncthreads();

    // y = attn . v  (f32 WMMA 16x16x4, K=NTOK, B from transposed copy)
    for (int dcb = 0; dcb < 2; ++dcb) {
        int dc = dcb * 16;
        v8f acc = {};
        #pragma unroll
        for (int kk = 0; kk < NTOK; kk += 4) {
            v2f af = *reinterpret_cast<const v2f*>(&at[tr + col][kk + 2 * half]);
            v2f bf = *reinterpret_cast<const v2f*>(&vsT[dc + col][kk + 2 * half]);
            acc = __builtin_amdgcn_wmma_f32_16x16x4_f32(false, af, false, bf,
                                                        (short)0, acc, false, false);
        }
        #pragma unroll
        for (int i = 0; i < 8; ++i) {
            int t = tr + i + 8 * half, d = dc + col;
            Y[(size_t)wi * NTOK * CC + t * CC + hd * HD + d] = acc[i];
        }
    }
}

// ---------------- kernel 3: output projection GEMM (bf16 WMMA) ----------------
__global__ __launch_bounds__(256)
void proj_gemm_kernel(const float* __restrict__ Yw, const float* __restrict__ pw,
                      const float* __restrict__ pb, float* __restrict__ out) {
    __shared__ __bf16 As[64][CC + 8];   // [m][k]; re-used as f32 C-staging after barrier
    __shared__ __bf16 Bs[64][CC + 8];   // [n][k]

    const int m0  = blockIdx.x * 64;    // == wi * 64
    const int n0  = blockIdx.y * 64;
    const int tid = threadIdx.x;

    for (int i = tid; i < 64 * CC / 4; i += 256) {
        int m = i / 48, c4 = (i % 48) * 4;
        float4 yv = *reinterpret_cast<const float4*>(&Yw[(size_t)(m0 + m) * CC + c4]);
        As[m][c4 + 0] = (__bf16)yv.x;
        As[m][c4 + 1] = (__bf16)yv.y;
        As[m][c4 + 2] = (__bf16)yv.z;
        As[m][c4 + 3] = (__bf16)yv.w;
    }
    for (int i = tid; i < 64 * CC / 4; i += 256) {
        int nl = i / 48, c4 = (i % 48) * 4;
        float4 wv = *reinterpret_cast<const float4*>(&pw[(n0 + nl) * CC + c4]);
        Bs[nl][c4 + 0] = (__bf16)wv.x;
        Bs[nl][c4 + 1] = (__bf16)wv.y;
        Bs[nl][c4 + 2] = (__bf16)wv.z;
        Bs[nl][c4 + 3] = (__bf16)wv.w;
    }
    __syncthreads();

    const int wave = tid >> 5, lane = tid & 31;
    const int wm  = (wave & 3) * 16;
    const int wnb = (wave >> 2) * 32;
    const int half = lane >> 4, l15 = lane & 15;

    const __bf16* arow  = As[wm + l15];
    const __bf16* brow0 = Bs[wnb + l15];
    const __bf16* brow1 = Bs[wnb + 16 + l15];

    v8f acc0 = {}, acc1 = {};
    #pragma unroll
    for (int kk = 0; kk < CC; kk += 32) {
        v16bf af  = load_frag_bf16(arow,  kk, half);
        v16bf bf0 = load_frag_bf16(brow0, kk, half);
        v16bf bf1 = load_frag_bf16(brow1, kk, half);
        acc0 = __builtin_amdgcn_wmma_f32_16x16x32_bf16(false, af, false, bf0,
                                                       (short)0, acc0, false, false);
        acc1 = __builtin_amdgcn_wmma_f32_16x16x32_bf16(false, af, false, bf1,
                                                       (short)0, acc1, false, false);
    }
    __syncthreads();   // all waves done reading As -> safe to alias as C staging

    float (*Cs)[66] = reinterpret_cast<float (*)[66]>(&As[0][0]);  // 64x66 f32 = 16.9KB
    #pragma unroll
    for (int s = 0; s < 2; ++s) {
        const v8f& acc = s ? acc1 : acc0;
        int ncol = wnb + 16 * s + l15;
        float bv = pb[n0 + ncol];
        #pragma unroll
        for (int i = 0; i < 8; ++i)
            Cs[wm + i + 8 * half][ncol] = acc[i] + bv;
    }
    __syncthreads();

    // window-reverse store: out[b, oc, h, w]; 2 rows x 8 contiguous w per thread
    const int wi = m0 >> 6;
    const int b  = wi >> 8;
    const int win = wi & 255;
    const int wy = win >> 4, wx = win & 15;
    const int ocl = tid >> 2;               // 0..63
    const int rr  = (tid & 3) * 2;          // token rows rr, rr+1
    const int oc  = n0 + ocl;
    #pragma unroll
    for (int rj = 0; rj < 2; ++rj) {
        int r = rr + rj;
        size_t obase = (((size_t)b * CC + oc) * HH + wy * 8 + r) * WW + wx * 8;
        float4 o0 = {Cs[r * 8 + 0][ocl], Cs[r * 8 + 1][ocl],
                     Cs[r * 8 + 2][ocl], Cs[r * 8 + 3][ocl]};
        float4 o1 = {Cs[r * 8 + 4][ocl], Cs[r * 8 + 5][ocl],
                     Cs[r * 8 + 6][ocl], Cs[r * 8 + 7][ocl]};
        *reinterpret_cast<float4*>(&out[obase])     = o0;
        *reinterpret_cast<float4*>(&out[obase + 4]) = o1;
    }
}

// ---------------- launcher ----------------
extern "C" void kernel_launch(void* const* d_in, const int* in_sizes, int n_in,
                              void* d_out, int out_size, void* d_ws, size_t ws_size,
                              hipStream_t stream) {
    (void)in_sizes; (void)n_in; (void)out_size; (void)ws_size;
    const float* x    = (const float*)d_in[0];
    const float* V_w  = (const float*)d_in[1];
    const float* V_b  = (const float*)d_in[2];
    const float* QK_w = (const float*)d_in[3];
    const float* QK_b = (const float*)d_in[4];
    const float* P_w  = (const float*)d_in[5];
    const float* P_b  = (const float*)d_in[6];
    const float* m_w1 = (const float*)d_in[7];
    const float* m_b1 = (const float*)d_in[8];
    const float* m_w2 = (const float*)d_in[9];
    const float* m_b2 = (const float*)d_in[10];
    float* out = (float*)d_out;

    float* ws = (float*)d_ws;
    const size_t QS = (size_t)NWIN * NTOK * CC;   // 25,165,824 floats per buffer
    float* Q    = ws;
    float* K    = ws + QS;
    float* V    = ws + 2 * QS;
    float* Y    = ws + 3 * QS;
    float* BIAS = ws + 4 * QS;                    // NHD*64*64 floats

    meta_bias_kernel<<<16, 256, 0, stream>>>(m_w1, m_b1, m_w2, m_b2, BIAS);
    qkv_gemm_kernel<<<dim3(BB * HWSZ / 64, 576 / 64), 256, 0, stream>>>(
        x, QK_w, QK_b, V_w, V_b, Q, K, V);
    attn_kernel<<<NWIN * NHD, 128, 0, stream>>>(Q, K, V, BIAS, Y);
    proj_gemm_kernel<<<dim3(NWIN, CC / 64), 256, 0, stream>>>(Y, P_w, P_b, out);
}
